// SimulationGCN_71605694759081
// MI455X (gfx1250) — compile-verified, matrix operand order checked by reading, not run on hardware
//
#include <hip/hip_runtime.h>

typedef __attribute__((ext_vector_type(16))) __bf16 v16bf;
typedef __attribute__((ext_vector_type(8)))  float  v8f;

#define NN  100000
#define EE  3200000
#define INF 128
#define HID 64
#define TILES (NN / 16)   // 6250 row tiles of 16

#define WMMA_BF16(A, B, C) \
  __builtin_amdgcn_wmma_f32_16x16x32_bf16(false, (A), false, (B), (short)0, (C), false, false)

// A-fragment (16x32 bf16, wave32): lane holds row m = lane&15, K-base
// kb = (lane>>4)*8; elements are p[0..7] and p[16..23] -> two contiguous
// 8-float runs, converted f32->bf16 (compiler emits v_cvt_pk_bf16_f32).
static __device__ __forceinline__ v16bf load_a16(const float* __restrict__ p) {
  v16bf a;
#pragma unroll
  for (int i = 0; i < 8; ++i) {
    a[i]     = (__bf16)p[i];
    a[i + 8] = (__bf16)p[i + 16];
  }
  return a;
}

// Pack W[K,N] (row-major f32) into per-lane WMMA B fragments in LDS:
// fragment (kkIdx, ct) holds the 32x16 bf16 tile; lane l's 16 elements are
// contiguous at wpack[frag*512 + l*16], so reads are 2x ds_load_b128.
static __device__ __forceinline__ void pack_w(__bf16* __restrict__ wpack,
                                              const float* __restrict__ W,
                                              int kTiles) {
  const int total = kTiles * (HID / 16) * 512;
  for (int idx = threadIdx.x; idx < total; idx += 256) {
    const int e     = idx & 15;
    const int l     = (idx >> 4) & 31;
    const int frag  = idx >> 9;
    const int kkIdx = frag >> 2;          // HID/16 == 4 col tiles
    const int ct    = frag & 3;
    const int n     = ct * 16 + (l & 15);
    const int kb    = (l >> 4) * 8;
    const int K     = kkIdx * 32 + ((e < 8) ? (kb + e) : (16 + kb + (e - 8)));
    wpack[idx] = (__bf16)W[K * HID + n];
  }
}

// ---- layer 1 GEMM: out[N,64] = x[N,128] @ W[128,64], bf16 WMMA, f32 acc ----
__global__ __launch_bounds__(256) void gemm1_wmma(const float* __restrict__ x,
                                                  const float* __restrict__ W,
                                                  float* __restrict__ out) {
  __shared__ alignas(32) __bf16 wpack[4 * 4 * 512];   // 16 KB
  pack_w(wpack, W, INF / 32);
  __syncthreads();

  const int lane = threadIdx.x & 31;
  const int wave = threadIdx.x >> 5;
  const v16bf* wfrag = (const v16bf*)wpack;
  v16bf B[4][4];                                      // hoisted: 128 VGPRs
#pragma unroll
  for (int k = 0; k < 4; ++k)
#pragma unroll
    for (int c = 0; c < 4; ++c)
      B[k][c] = wfrag[(k * 4 + c) * 32 + lane];

  const int m    = lane & 15;
  const int kb   = (lane >> 4) * 8;
  const int step = gridDim.x * 8;
  for (int tile = blockIdx.x * 8 + wave; tile < TILES; tile += step) {
    const int rowBase = tile * 16;
    const float* arow = x + (size_t)(rowBase + m) * INF + kb;
    if (tile + step < TILES)
      __builtin_prefetch(x + (size_t)((tile + step) * 16 + m) * INF + kb, 0, 0);
    v8f c0 = {}, c1 = {}, c2 = {}, c3 = {};
#pragma unroll
    for (int k = 0; k < 4; ++k) {
      v16bf a = load_a16(arow + k * 32);
      c0 = WMMA_BF16(a, B[k][0], c0);
      c1 = WMMA_BF16(a, B[k][1], c1);
      c2 = WMMA_BF16(a, B[k][2], c2);
      c3 = WMMA_BF16(a, B[k][3], c3);
    }
    float* orow = out + (size_t)(rowBase + (lane >> 4) * 8) * HID + m;
#pragma unroll
    for (int r = 0; r < 8; ++r) {
      orow[r * HID +  0] = c0[r];
      orow[r * HID + 16] = c1[r];
      orow[r * HID + 32] = c2[r];
      orow[r * HID + 48] = c3[r];
    }
  }
}

// ---- layer 2 GEMM, bias+ReLU fused on A: out = relu(hin+b) @ W ----
__global__ __launch_bounds__(256) void gemm2_wmma(const float* __restrict__ hin,
                                                  const float* __restrict__ W,
                                                  const float* __restrict__ bias,
                                                  float* __restrict__ out) {
  __shared__ alignas(32) __bf16 wpack[2 * 4 * 512];   // 8 KB
  pack_w(wpack, W, HID / 32);
  __syncthreads();

  const int lane = threadIdx.x & 31;
  const int wave = threadIdx.x >> 5;
  const int m    = lane & 15;
  const int kb   = (lane >> 4) * 8;

  const v16bf* wfrag = (const v16bf*)wpack;
  v16bf B[2][4];                                      // hoisted: 64 VGPRs
#pragma unroll
  for (int k = 0; k < 2; ++k)
#pragma unroll
    for (int c = 0; c < 4; ++c)
      B[k][c] = wfrag[(k * 4 + c) * 32 + lane];

  float ba[2][16];                                    // per-lane bias, hoisted
#pragma unroll
  for (int k = 0; k < 2; ++k)
#pragma unroll
    for (int i = 0; i < 8; ++i) {
      ba[k][i]     = bias[k * 32 + kb + i];
      ba[k][i + 8] = bias[k * 32 + 16 + kb + i];
    }

  const int step = gridDim.x * 8;
  for (int tile = blockIdx.x * 8 + wave; tile < TILES; tile += step) {
    const int rowBase = tile * 16;
    const float* arow = hin + (size_t)(rowBase + m) * HID + kb;
    if (tile + step < TILES)
      __builtin_prefetch(hin + (size_t)((tile + step) * 16 + m) * HID + kb, 0, 0);
    v8f c0 = {}, c1 = {}, c2 = {}, c3 = {};
#pragma unroll
    for (int k = 0; k < 2; ++k) {
      const float* p = arow + k * 32;
      v16bf a;
#pragma unroll
      for (int i = 0; i < 8; ++i) {
        a[i]     = (__bf16)fmaxf(p[i]      + ba[k][i],     0.0f);
        a[i + 8] = (__bf16)fmaxf(p[i + 16] + ba[k][i + 8], 0.0f);
      }
      c0 = WMMA_BF16(a, B[k][0], c0);
      c1 = WMMA_BF16(a, B[k][1], c1);
      c2 = WMMA_BF16(a, B[k][2], c2);
      c3 = WMMA_BF16(a, B[k][3], c3);
    }
    float* orow = out + (size_t)(rowBase + (lane >> 4) * 8) * HID + m;
#pragma unroll
    for (int r = 0; r < 8; ++r) {
      orow[r * HID +  0] = c0[r];
      orow[r * HID + 16] = c1[r];
      orow[r * HID + 32] = c2[r];
      orow[r * HID + 48] = c3[r];
    }
  }
}

// ---- SpMM scatter: acc[row[e],:] += val[e] * feat[col[e],:] ----
// 32 threads per edge; lane handles features f and f+32.
__global__ __launch_bounds__(256) void spmm_scatter(const int* __restrict__ erow,
                                                    const int* __restrict__ ecol,
                                                    const float* __restrict__ ev,
                                                    const float* __restrict__ feat,
                                                    float* __restrict__ acc) {
  long gid = (long)blockIdx.x * 256 + threadIdx.x;
  int e = (int)(gid >> 5);
  if (e >= EE) return;
  int f = (int)(gid & 31);
  int r = erow[e];
  int c = ecol[e];
  float v = ev[e];
  const float* fp = feat + (size_t)c * HID + f;
  float*       ap = acc  + (size_t)r * HID + f;
  atomicAdd(ap,      v * fp[0]);
  atomicAdd(ap + 32, v * fp[32]);
}

// ---- final: out[n] = relu(h[n]+b2) . lin_W + lin_b (one wave per node) ----
__global__ __launch_bounds__(256) void final_linear(const float* __restrict__ h,
                                                    const float* __restrict__ b2,
                                                    const float* __restrict__ lw,
                                                    const float* __restrict__ lb,
                                                    float* __restrict__ out) {
  int lane = threadIdx.x & 31;
  int wave = threadIdx.x >> 5;
  int node = blockIdx.x * 8 + wave;
  if (node >= NN) return;
  const float* hr = h + (size_t)node * HID + lane * 2;
  float x0 = fmaxf(hr[0] + b2[lane * 2 + 0], 0.0f);
  float x1 = fmaxf(hr[1] + b2[lane * 2 + 1], 0.0f);
  float part = x0 * lw[lane * 2] + x1 * lw[lane * 2 + 1];
#pragma unroll
  for (int off = 16; off > 0; off >>= 1)
    part += __shfl_down(part, off, 32);
  if (lane == 0) out[node] = part + lb[0];
}

__global__ void zero_kernel(float4* __restrict__ p, int n4) {
  int i = blockIdx.x * blockDim.x + threadIdx.x;
  if (i < n4) p[i] = make_float4(0.f, 0.f, 0.f, 0.f);
}

extern "C" void kernel_launch(void* const* d_in, const int* in_sizes, int n_in,
                              void* d_out, int out_size, void* d_ws, size_t ws_size,
                              hipStream_t stream) {
  const float* x   = (const float*)d_in[0];
  const int*   er  = (const int*)  d_in[1];
  const int*   ec  = (const int*)  d_in[2];
  const float* ev  = (const float*)d_in[3];
  const float* W1  = (const float*)d_in[4];
  const float* b1  = (const float*)d_in[5];
  const float* W2  = (const float*)d_in[6];
  const float* b2  = (const float*)d_in[7];
  const float* lw  = (const float*)d_in[8];
  const float* lb  = (const float*)d_in[9];
  float* out = (float*)d_out;

  const size_t bufElems = (size_t)NN * HID;         // 6.4M floats = 25.6 MB
  float* bufA = (float*)d_ws;                       // GEMM outputs (reused)
  float* bufB = bufA + bufElems;                    // agg layer 1
  float* bufC = bufB + bufElems;                    // agg layer 2

  const int n4 = (int)(bufElems / 4);
  zero_kernel<<<(n4 + 255) / 256, 256, 0, stream>>>((float4*)bufB, n4);
  zero_kernel<<<(n4 + 255) / 256, 256, 0, stream>>>((float4*)bufC, n4);

  const unsigned gemmGrid = 192;                    // persistent waves
  const long scatterThreads = (long)EE * 32;        // 102.4M
  const unsigned scatterGrid = (unsigned)((scatterThreads + 255) / 256);

  gemm1_wmma<<<gemmGrid, 256, 0, stream>>>(x, W1, bufA);
  spmm_scatter<<<scatterGrid, 256, 0, stream>>>(er, ec, ev, bufA, bufB);
  gemm2_wmma<<<gemmGrid, 256, 0, stream>>>(bufB, W2, b1, bufA);
  spmm_scatter<<<scatterGrid, 256, 0, stream>>>(er, ec, ev, bufA, bufC);
  final_linear<<<(NN + 7) / 8, 256, 0, stream>>>(bufC, b2, lw, lb, out);
}
